// TransformerEncoderLayerWithMixedAttention_65077344469580
// MI455X (gfx1250) — matrix operand order repaired and checked by reading, not compiled, not run on hardware
//
#include <hip/hip_runtime.h>
#include <cstddef>
#include <cstdint>

// ---------------- types & helpers ----------------
typedef __bf16 bf16_t;
typedef __attribute__((ext_vector_type(16))) __bf16 v16bf;
typedef __attribute__((ext_vector_type(8)))  __bf16 v8bf;
typedef __attribute__((ext_vector_type(8)))  float  v8f;
typedef __attribute__((ext_vector_type(4)))  int    v4i;

// gfx1250 async global->LDS path (guarded: falls back to manual staging if the
// toolchain does not expose the builtins)
#if defined(__AMDGCN__) && __has_builtin(__builtin_amdgcn_global_load_async_to_lds_b128) && __has_builtin(__builtin_amdgcn_s_wait_asynccnt)
#define HAVE_ASYNC_LDS 1
typedef __attribute__((address_space(1))) v4i* as1_v4i_p;
typedef __attribute__((address_space(3))) v4i* as3_v4i_p;
typedef __attribute__((address_space(3))) char* as3char_p;
#else
#define HAVE_ASYNC_LDS 0
#endif

__device__ __forceinline__ bf16_t f2bf(float f) {
  unsigned u; __builtin_memcpy(&u, &f, 4);
  unsigned r = (u + 0x7FFFu + ((u >> 16) & 1u)) >> 16;
  unsigned short s = (unsigned short)r;
  bf16_t h; __builtin_memcpy(&h, &s, 2); return h;
}
__device__ __forceinline__ v8bf ld8(const bf16_t* p) {
  v8bf r; __builtin_memcpy(&r, p, sizeof(r)); return r;
}
__device__ __forceinline__ v16bf mk16(v8bf a, v8bf b) {
  return __builtin_shufflevector(a, b, 0,1,2,3,4,5,6,7,8,9,10,11,12,13,14,15);
}
// A-operand (16x32, MxK): lane holds row M=lane&15; lane<16 -> K 0..15, lane>=16 -> K 16..31 (contiguous 16)
__device__ __forceinline__ v16bf ldA(const bf16_t* rowp, int lane) {
  int off = (lane >> 4) ? 16 : 0;
  return mk16(ld8(rowp + off), ld8(rowp + off + 8));
}
// B-operand (32x16, KxN): lane holds col N=lane&15; lane<16 -> K 0..15, lane>=16 -> K 16..31 (contiguous 16)
__device__ __forceinline__ v16bf ldB(const bf16_t* rowp, int lane) {
  int off = (lane >> 4) ? 16 : 0;
  return mk16(ld8(rowp + off), ld8(rowp + off + 8));
}
__device__ __forceinline__ v8f wmma_bf16(v16bf a, v16bf b, v8f c) {
  return __builtin_amdgcn_wmma_f32_16x16x32_bf16(false, a, false, b, (short)0, c, false, false);
}

// ---------------- f32 -> bf16 convert ----------------
__global__ void f2b_kernel(const float* __restrict__ in, bf16_t* __restrict__ out, size_t n) {
  size_t i = blockIdx.x * (size_t)blockDim.x + threadIdx.x;
  size_t stride = (size_t)gridDim.x * blockDim.x;
  for (; i < n; i += stride) out[i] = f2bf(in[i]);
}

// ---------------- x + pos (pos indexed by batch — reference quirk) ----------------
__global__ void addpos_kernel(const float* __restrict__ x, const float* __restrict__ pos,
                              float* __restrict__ xres, bf16_t* __restrict__ xb,
                              int B, int E, size_t total) {
  size_t i = blockIdx.x * (size_t)blockDim.x + threadIdx.x;
  if (i >= total) return;
  size_t t = i / (size_t)E;
  int c = (int)(i - t * (size_t)E);
  int b = (int)(t % (size_t)B);
  float v = x[i] + pos[(size_t)b * E + c];
  xres[i] = v;
  xb[i] = f2bf(v);
}

// ---------------- generic bf16 WMMA GEMM: C(M,N) = A(M,K) @ W(N,K)^T + bias ----------------
// 32(M) x 64(N) tile per wave, K swept in 64-element ping-pong stages so loads
// land directly in the registers the WMMAs consume (no rotation copies) and
// sixteen WMMAs issue per loop iteration. Requires K % 64 == 0, M % 32 == 0.
__global__ __launch_bounds__(128)
void gemm_kernel(const bf16_t* __restrict__ A, const bf16_t* __restrict__ W,
                 const float* __restrict__ bias, float* __restrict__ outF,
                 bf16_t* __restrict__ outB, int M, int N, int K, int relu) {
  const int lane = threadIdx.x & 31;
  const int wave = threadIdx.x >> 5;
  const int mbase = blockIdx.x * 32;
  const int nbase = blockIdx.y * 256 + wave * 64;
  if (mbase >= M || nbase >= N) return;

  const int m = lane & 15;
  const bf16_t* arow0 = A + (size_t)(mbase + m) * K;        // rows 0..15 of tile
  const bf16_t* arow1 = A + (size_t)(mbase + 16 + m) * K;   // rows 16..31 of tile
  const bf16_t* wrow[4];
  for (int t = 0; t < 4; ++t)
    wrow[t] = W + (size_t)(nbase + t * 16 + (lane & 15)) * K;

  v8f acc[2][4];
  for (int t = 0; t < 4; ++t) {
    float bv = bias ? bias[nbase + t * 16 + (lane & 15)] : 0.0f;
    for (int r = 0; r < 8; ++r) { acc[0][t][r] = bv; acc[1][t][r] = bv; }
  }

  // stage-0 prologue
  v16bf aS0[2], bS0[4], aS1[2], bS1[4];
  aS0[0] = ldA(arow0, lane);
  aS0[1] = ldA(arow1, lane);
  for (int t = 0; t < 4; ++t) bS0[t] = ldB(wrow[t], lane);

  for (int k0 = 0; k0 < K; k0 += 64) {
    // fetch stage 1 (k0+32)
    aS1[0] = ldA(arow0 + k0 + 32, lane);
    aS1[1] = ldA(arow1 + k0 + 32, lane);
    for (int t = 0; t < 4; ++t) bS1[t] = ldB(wrow[t] + k0 + 32, lane);
    __builtin_prefetch(arow0 + k0 + 512, 0, 1);
    // compute stage 0
    for (int t = 0; t < 4; ++t) {
      acc[0][t] = wmma_bf16(aS0[0], bS0[t], acc[0][t]);
      acc[1][t] = wmma_bf16(aS0[1], bS0[t], acc[1][t]);
    }
    // fetch stage 0 of next iteration (k0+64)
    if (k0 + 64 < K) {
      aS0[0] = ldA(arow0 + k0 + 64, lane);
      aS0[1] = ldA(arow1 + k0 + 64, lane);
      for (int t = 0; t < 4; ++t) bS0[t] = ldB(wrow[t] + k0 + 64, lane);
    }
    // compute stage 1
    for (int t = 0; t < 4; ++t) {
      acc[0][t] = wmma_bf16(aS1[0], bS1[t], acc[0][t]);
      acc[1][t] = wmma_bf16(aS1[1], bS1[t], acc[1][t]);
    }
  }

  const int halfoff = (lane >> 4) ? 8 : 0;
  for (int half = 0; half < 2; ++half)
    for (int t = 0; t < 4; ++t)
      for (int r = 0; r < 8; ++r) {
        float v = acc[half][t][r];
        if (relu) v = fmaxf(v, 0.0f);
        size_t row = (size_t)(mbase + half * 16 + halfoff + r);
        size_t col = (size_t)(nbase + t * 16 + (lane & 15));
        if (outF) outF[row * N + col] = v;
        if (outB) outB[row * N + col] = f2bf(v);
      }
}

// ---------------- masked flash attention: one wave per (16-query tile, b, h) ----------------
#define ATT_MAXT 28
#define ATT_LDC  448   // ATT_MAXT * 16

__global__ __launch_bounds__(32)
void attn_kernel(const bf16_t* __restrict__ qkv, bf16_t* __restrict__ y,
                 const int* __restrict__ lwsp, int S, int B, int H, int E) {
  const int hd = 64;
  const int lane = threadIdx.x;
  const int qt = blockIdx.x / (B * H);
  const int bh = blockIdx.x % (B * H);
  const int b = bh / H, h = bh % H;
  const int q0 = qt * 16;
  int w = lwsp[0];
  int tw = 2 * w; if (tw < 1) tw = 1;
  const int threeE = 3 * E;

  __shared__ float  s_sc[16 * ATT_LDC];
  __shared__ bf16_t s_p[16 * ATT_LDC];
  __shared__ bf16_t s_v[64 * 32];       // transposed: [hd col][key]
#if HAVE_ASYNC_LDS
  __shared__ bf16_t s_vrm[32 * 64];     // row-major async landing: [key][hd col]
#endif
  __shared__ int    s_tj[ATT_MAXT + 4];

  const int m = lane & 15;
  const int n = lane & 15;
  const bf16_t* qrow = qkv + (size_t)((q0 + m) * B + b) * threeE + h * hd;
  v16bf aq0 = ldA(qrow + 0, lane);
  v16bf aq1 = ldA(qrow + 32, lane);
  const float scale = rsqrtf((float)hd);

  // pass 1: scores for allowed k-tiles, masked, stashed in LDS
  int ntiles = 0;
  const int nkt = S / 16;
  for (int kt = 0; kt < nkt; ++kt) {
    int j0 = kt * 16;
    bool win = (j0 + 15 >= q0 - w) && (j0 <= q0 + 15 + w);
    int fm = ((j0 + tw - 1) / tw) * tw;   // first multiple of 2w >= j0
    bool glob = (fm <= j0 + 15);
    if (!(win || glob)) continue;
    if (ntiles >= ATT_MAXT) break;

    const bf16_t* krow = qkv + (size_t)((j0 + n) * B + b) * threeE + E + h * hd;
    v16bf bk0 = ldB(krow + 0, lane);
    v16bf bk1 = ldB(krow + 32, lane);
    v8f s;
    for (int r = 0; r < 8; ++r) s[r] = 0.0f;
    s = wmma_bf16(aq0, bk0, s);
    s = wmma_bf16(aq1, bk1, s);

    int halfoff = (lane >> 4) ? 8 : 0;
    for (int r = 0; r < 8; ++r) {
      int mi = halfoff + r;
      int i = q0 + mi;
      int j = j0 + n;
      int d = (i > j) ? (i - j) : (j - i);
      bool ok = (d <= w) || (j % tw == 0);
      s_sc[mi * ATT_LDC + ntiles * 16 + n] = ok ? s[r] * scale : -__builtin_inff();
    }
    if (lane == 0) s_tj[ntiles] = j0;
    ++ntiles;
  }
  __syncthreads();

  // pass 2: row softmax (lanes split columns by parity; combine across lane-halves)
  const int ncols = ntiles * 16;
  const int row = lane & 15;
  const int c0 = (lane >> 4);
  float mx = -__builtin_inff();
  for (int c = c0; c < ncols; c += 2) mx = fmaxf(mx, s_sc[row * ATT_LDC + c]);
  mx = fmaxf(mx, __shfl_xor(mx, 16, 32));
  float sum = 0.0f;
  for (int c = c0; c < ncols; c += 2) {
    float e = __expf(s_sc[row * ATT_LDC + c] - mx);
    s_sc[row * ATT_LDC + c] = e;
    sum += e;
  }
  sum += __shfl_xor(sum, 16, 32);
  float inv = 1.0f / sum;
  for (int c = c0; c < ncols; c += 2)
    s_p[row * ATT_LDC + c] = f2bf(s_sc[row * ATT_LDC + c] * inv);
  if (ntiles & 1) {  // zero-pad one tile so P@V can run K=32 chunks
    for (int c = c0; c < 16; c += 2) s_p[row * ATT_LDC + ncols + c] = f2bf(0.0f);
  }
  const int ptiles = (ntiles + 1) & ~1;
  __syncthreads();

  // pass 3: y += P @ V, 32 keys per step
  v8f acc[4];
  for (int t = 0; t < 4; ++t)
    for (int r = 0; r < 8; ++r) acc[t][r] = 0.0f;

  for (int p = 0; p < ptiles; p += 2) {
#if HAVE_ASYNC_LDS
    {
      // fence: previous iteration's DS reads (DScnt) are unordered vs ASYNC LDS writes
      asm volatile("s_wait_dscnt 0" ::: "memory");
      // async-copy 32 key rows (128B each) row-major into s_vrm:
      // per instruction: 32 lanes x 16B = 4 key rows; 8 instructions move the tile
      const int kk4 = lane >> 3;            // 0..3: key-within-group
      const int cb  = (lane & 7) * 16;      // byte chunk within 128B row
      as3char_p ldsbase = (as3char_p)(void*)&s_vrm[0];
      for (int i = 0; i < 8; ++i) {
        int key = i * 4 + kk4;
        int ti = p + (key >> 4);
        int j0t = (ti < ntiles) ? s_tj[ti] : 0;   // padded tile: P==0, any valid row
        int j = j0t + (key & 15);
        const bf16_t* vr = qkv + (size_t)(j * B + b) * threeE + 2 * E + h * hd;
        as1_v4i_p gsrc = (as1_v4i_p)((uintptr_t)vr + (uintptr_t)cb);
        as3_v4i_p ldst = (as3_v4i_p)(ldsbase + key * 128 + cb);
        __builtin_amdgcn_global_load_async_to_lds_b128(gsrc, ldst, 0, 0);
      }
      __builtin_amdgcn_s_wait_asynccnt(0);
      // transpose in LDS: each lane owns key row `lane`, vectorized reads
      const bf16_t* myrow = &s_vrm[0] + lane * 64;
      for (int c8 = 0; c8 < 8; ++c8) {
        v8bf vv = ld8(myrow + c8 * 8);
        for (int e = 0; e < 8; ++e) s_v[(c8 * 8 + e) * 32 + lane] = vv[e];
      }
    }
#else
    // fallback: gather V directly from global, storing transposed
    for (int kk = 0; kk < 32; ++kk) {
      int ti = p + (kk >> 4);
      int j0t = (ti < ntiles) ? s_tj[ti] : 0;
      int j = j0t + (kk & 15);
      const bf16_t* vr = qkv + (size_t)(j * B + b) * threeE + 2 * E + h * hd;
      s_v[lane * 32 + kk] = vr[lane];
      s_v[(lane + 32) * 32 + kk] = vr[lane + 32];
    }
#endif
    __syncthreads();
    const bf16_t* prow = s_p + row * ATT_LDC + p * 16;
    v16bf pa = ldA(prow, lane);
    for (int t = 0; t < 4; ++t) {
      const bf16_t* vb = s_v + (size_t)(t * 16 + (lane & 15)) * 32 + ((lane >> 4) ? 16 : 0);
      v16bf bt = mk16(ld8(vb), ld8(vb + 8));
      acc[t] = wmma_bf16(pa, bt, acc[t]);
    }
    __syncthreads();
  }

  // write y tile (bf16, consumed by out-proj GEMM)
  const int halfoff = (lane >> 4) ? 8 : 0;
  for (int t = 0; t < 4; ++t)
    for (int r = 0; r < 8; ++r) {
      int mi = halfoff + r;
      size_t trow = (size_t)((q0 + mi) * B + b);
      size_t col = (size_t)(h * hd + t * 16 + (lane & 15));
      y[trow * E + col] = f2bf(acc[t][r]);
    }
}

// ---------------- residual + LayerNorm (E==512, 256 threads, 2 elems/thread) ----------------
__global__ __launch_bounds__(256)
void ln_kernel(const float* __restrict__ a, const float* __restrict__ rsd,
               const float* __restrict__ g, const float* __restrict__ be,
               float* __restrict__ outF, bf16_t* __restrict__ outB, int E) {
  __shared__ float red[256];
  const int row = blockIdx.x;
  const int tid = threadIdx.x;
  const float* pa = a + (size_t)row * E;
  const float* pr = rsd + (size_t)row * E;
  float v0 = pa[tid] + pr[tid];
  float v1 = pa[tid + 256] + pr[tid + 256];

  red[tid] = v0 + v1; __syncthreads();
  for (int o = 128; o > 0; o >>= 1) { if (tid < o) red[tid] += red[tid + o]; __syncthreads(); }
  float mu = red[0] / (float)E; __syncthreads();

  float d0 = v0 - mu, d1 = v1 - mu;
  red[tid] = d0 * d0 + d1 * d1; __syncthreads();
  for (int o = 128; o > 0; o >>= 1) { if (tid < o) red[tid] += red[tid + o]; __syncthreads(); }
  float rstd = rsqrtf(red[0] / (float)E + 1e-5f);

  float o0 = d0 * rstd * g[tid] + be[tid];
  float o1 = d1 * rstd * g[tid + 256] + be[tid + 256];
  if (outF) { float* po = outF + (size_t)row * E; po[tid] = o0; po[tid + 256] = o1; }
  if (outB) { bf16_t* pb = outB + (size_t)row * E; pb[tid] = f2bf(o0); pb[tid + 256] = f2bf(o1); }
}

// ---------------- launcher ----------------
extern "C" void kernel_launch(void* const* d_in, const int* in_sizes, int n_in,
                              void* d_out, int out_size, void* d_ws, size_t ws_size,
                              hipStream_t stream) {
  (void)in_sizes; (void)n_in; (void)out_size; (void)ws_size;
  const int S = 2048, B = 4, E = 512, H = 8, FF = 2048;
  const int M = S * B;                 // 8192 token rows, layout (S,B,E) flattened

  const float* x     = (const float*)d_in[0];
  const float* pos   = (const float*)d_in[1];
  const float* in_w  = (const float*)d_in[2];
  const float* in_b  = (const float*)d_in[3];
  const float* out_w = (const float*)d_in[4];
  const float* out_b = (const float*)d_in[5];
  const float* w1    = (const float*)d_in[6];
  const float* b1    = (const float*)d_in[7];
  const float* w2    = (const float*)d_in[8];
  const float* b2    = (const float*)d_in[9];
  const float* g1    = (const float*)d_in[10];
  const float* be1   = (const float*)d_in[11];
  const float* g2    = (const float*)d_in[12];
  const float* be2   = (const float*)d_in[13];
  const int*   lws   = (const int*)d_in[14];
  float* out = (float*)d_out;

  char* p = (char*)d_ws;
  auto alloc = [&](size_t bytes) { char* r = p; p += (bytes + 255) & ~(size_t)255; return r; };
  bf16_t* xb    = (bf16_t*)alloc((size_t)M * E * 2);
  float*  xres  = (float*) alloc((size_t)M * E * 4);
  bf16_t* qkvb  = (bf16_t*)alloc((size_t)M * 3 * E * 2);   // 25.2 MB
  bf16_t* yb    = (bf16_t*)alloc((size_t)M * E * 2);       // contiguous after qkvb
  bf16_t* ff1b  = qkvb;                                    // alias: 33.5 MB == qkv+y, dead by FFN1
  float*  attnf = (float*) alloc((size_t)M * E * 4);
  float*  ff2f  = attnf;                                   // alias: attnf dead after LN1
  float*  x1f   = (float*) alloc((size_t)M * E * 4);
  bf16_t* x1b   = (bf16_t*)alloc((size_t)M * E * 2);
  bf16_t* wqkvb = (bf16_t*)alloc((size_t)3 * E * E * 2);
  bf16_t* woutb = (bf16_t*)alloc((size_t)E * E * 2);
  bf16_t* w1b   = (bf16_t*)alloc((size_t)FF * E * 2);
  bf16_t* w2b   = (bf16_t*)alloc((size_t)E * FF * 2);

  // weights -> bf16
  f2b_kernel<<<1024, 256, 0, stream>>>(in_w,  wqkvb, (size_t)3 * E * E);
  f2b_kernel<<<1024, 256, 0, stream>>>(out_w, woutb, (size_t)E * E);
  f2b_kernel<<<1024, 256, 0, stream>>>(w1,    w1b,   (size_t)FF * E);
  f2b_kernel<<<1024, 256, 0, stream>>>(w2,    w2b,   (size_t)E * FF);

  // x + pos (batch-indexed pos quirk)
  {
    size_t total = (size_t)M * E;
    addpos_kernel<<<(unsigned)((total + 255) / 256), 256, 0, stream>>>(x, pos, xres, xb, B, E, total);
  }

  // QKV projection: (M, 3E)
  gemm_kernel<<<dim3(M / 32, (3 * E) / 256), 128, 0, stream>>>(
      xb, wqkvb, in_b, nullptr, qkvb, M, 3 * E, E, 0);

  // masked attention
  attn_kernel<<<dim3((S / 16) * B * H), 32, 0, stream>>>(qkvb, yb, lws, S, B, H, E);

  // output projection
  gemm_kernel<<<dim3(M / 32, E / 256), 128, 0, stream>>>(
      yb, woutb, out_b, attnf, nullptr, M, E, E, 0);

  // LN1: x1 = LN(xres + attn_out)
  ln_kernel<<<M, 256, 0, stream>>>(xres, attnf, g1, be1, x1f, x1b, E);

  // FFN
  gemm_kernel<<<dim3(M / 32, FF / 256), 128, 0, stream>>>(
      x1b, w1b, b1, nullptr, ff1b, M, FF, E, 1);
  gemm_kernel<<<dim3(M / 32, E / 256), 128, 0, stream>>>(
      ff1b, w2b, b2, ff2f, nullptr, M, E, FF, 0);

  // LN2 -> output
  ln_kernel<<<M, 256, 0, stream>>>(x1f, ff2f, g2, be2, out, nullptr, E);
}
